// SoftDecisionTree_50062138802270
// MI455X (gfx1250) — matrix operand order repaired and checked by reading, not compiled, not run on hardware
//
#include <hip/hip_runtime.h>

// Fused soft-decision-tree forward for MI455X (gfx1250, wave32).
// Round 5: quad-buffered B fragments via a constant-indexed array (no rotation
// copies for the allocator to coalesce; 8 tiles/k-iter keeps the phase fixed).
// Load for tile t+2 targets the buffer consumed at t-2, so >=5 independent
// instructions separate the WMMA read from the overwriting load, satisfying
// the 4-coexec-slot WMMA->VALU WAR hazard (ISA 7.12.1) without v_nops.

typedef __attribute__((ext_vector_type(16))) _Float16 v16h;
typedef __attribute__((ext_vector_type(8)))  float    v8f;

namespace {

constexpr int kBatch    = 16384;
constexpr int kInDim    = 4096;
constexpr int kInternal = 1023;   // 2^10 - 1
constexpr int kLeaves   = 1024;
constexpr int kMTile    = 32;     // batch rows per block

union V16 { v16h v; _Float16 h[16]; float4 f4[2]; };

__device__ __forceinline__ void cvt8(const float* __restrict__ p, _Float16* __restrict__ h) {
  float4 a = *reinterpret_cast<const float4*>(p);
  float4 b = *reinterpret_cast<const float4*>(p + 4);
  h[0] = (_Float16)a.x; h[1] = (_Float16)a.y; h[2] = (_Float16)a.z; h[3] = (_Float16)a.w;
  h[4] = (_Float16)b.x; h[5] = (_Float16)b.y; h[6] = (_Float16)b.z; h[7] = (_Float16)b.w;
}

// A fragment (16x32 f16), own K order matched by B: lane L -> row (L&15);
// halves 0..15 = K kb..kb+15 contiguous, kb = (L>>4)*16.
__device__ __forceinline__ v16h load_a_frag(const float* __restrict__ x, unsigned off) {
  V16 u;
  cvt8(x + off,      u.h);
  cvt8(x + off + 8,  u.h + 8);
  return u.v;
}

// B fragment (32x16 f16) from pre-converted f16 weights: lane L -> col (L&15);
// halves 0..15 = K kb..kb+15 contiguous, kb = (L>>4)*16.  Two b128 loads, no VALU.
__device__ __forceinline__ v16h load_b_f16(const unsigned short* __restrict__ wh, unsigned off) {
  V16 u;
  u.f4[0] = *reinterpret_cast<const float4*>(wh + off);
  u.f4[1] = *reinterpret_cast<const float4*>(wh + off + 8);
  return u.v;
}

// Fallback B fragment straight from f32 weights (in-kernel convert).
__device__ __forceinline__ v16h load_b_f32(const float* __restrict__ w, unsigned off) {
  V16 u;
  cvt8(w + off,     u.h);
  cvt8(w + off + 8, u.h + 8);
  return u.v;
}

__device__ __forceinline__ v8f wmma16(v16h a, v16h b, v8f c) {
  return __builtin_amdgcn_wmma_f32_16x16x32_f16(
      /*neg_a=*/false, a, /*neg_b=*/false, b,
      /*c_mod=*/(short)0, c, /*reuse_a=*/false, /*reuse_b=*/false);
}

} // namespace

// ---- Pre-pass: W f32 -> f16 into workspace (run once per launch) ----------
__global__ __launch_bounds__(256)
void convert_w_f16(const float* __restrict__ w, unsigned short* __restrict__ wh) {
  int i = (blockIdx.x * blockDim.x + threadIdx.x) * 8;   // 8 elements / thread
  float4 a = *reinterpret_cast<const float4*>(w + i);
  float4 b = *reinterpret_cast<const float4*>(w + i + 4);
  V16 u;   // use low 8 halves
  u.h[0] = (_Float16)a.x; u.h[1] = (_Float16)a.y; u.h[2] = (_Float16)a.z; u.h[3] = (_Float16)a.w;
  u.h[4] = (_Float16)b.x; u.h[5] = (_Float16)b.y; u.h[6] = (_Float16)b.z; u.h[7] = (_Float16)b.w;
  *reinterpret_cast<float4*>(wh + i) = u.f4[0];
}

// ---- Fused GEMM + sigmoid + tree + class contraction ----------------------
template <bool WF16>
__global__ __launch_bounds__(256)
void sdt_fused_kernel(const float* __restrict__ x,
                      const float* __restrict__ w,
                      const unsigned short* __restrict__ wh,
                      const float* __restrict__ bias,
                      const float* __restrict__ leaf_logits,
                      float* __restrict__ out) {
  __shared__ _Float16 s_probs[kMTile * 1024];   // 64 KB split-probs (f16)
  __shared__ float    s_red[kMTile * 8 * 2];    // 2 KB partial sums

  const int tid  = threadIdx.x;
  const int lane = tid & 31;
  const int wave = tid >> 5;             // 0..7, each wave owns 128 output cols
  const int m0   = blockIdx.x * kMTile;

  const int nl = lane & 15;
  const int kh = lane >> 4;              // which K-half of the fragment

  // ---- GEMM phase: logits[32 x 1024] = x_tile @ W^T --------------------
  const unsigned a_kb  = (unsigned)(kh * 16);
  const unsigned b_kb  = (unsigned)(kh * 16);
  const unsigned aoff0 = (unsigned)(m0 + nl) * kInDim + a_kb;
  const unsigned aoff1 = (unsigned)(m0 + 16 + nl) * kInDim + a_kb;

  unsigned boff[8];
  int ncol[8];
#pragma unroll
  for (int t = 0; t < 8; ++t) {
    ncol[t] = wave * 128 + t * 16 + nl;
    int nc = ncol[t] > (kInternal - 1) ? (kInternal - 1) : ncol[t];  // clamp col 1023
    boff[t] = (unsigned)nc * kInDim + b_kb;
  }

  v8f acc[16] = {};                      // [0..7]: rows 0-15, [8..15]: rows 16-31

  // Quad-buffered B, constant indices: consume b[t&3], load 2 tiles ahead
  // into b[(t+2)&3] (the buffer consumed at tile t-2).
  v16h b[4];
  b[0] = WF16 ? load_b_f16(wh, boff[0]) : load_b_f32(w, boff[0]);
  b[1] = WF16 ? load_b_f16(wh, boff[1]) : load_b_f32(w, boff[1]);

  const int ktail = kInDim - 32;
  for (int k0 = 0; k0 < kInDim; k0 += 32) {
    __builtin_prefetch(x + aoff0 + k0 + 256, 0, 0);   // global_prefetch_b8
    __builtin_prefetch(x + aoff1 + k0 + 256, 0, 0);
    v16h a0 = load_a_frag(x, aoff0 + (unsigned)k0);
    v16h a1 = load_a_frag(x, aoff1 + (unsigned)k0);
    const int kn = (k0 == ktail) ? k0 : k0 + 32;      // last iter: dummy reload
#pragma unroll
    for (int t = 0; t < 8; ++t) {
      const int ts = t + 2;                           // tile loaded 2 ahead
      const unsigned offn = (ts < 8) ? (boff[ts]     + (unsigned)k0)
                                     : (boff[ts - 8] + (unsigned)kn);
      b[(t + 2) & 3] = WF16 ? load_b_f16(wh, offn) : load_b_f32(w, offn);
      acc[t]     = wmma16(a0, b[t & 3], acc[t]);
      acc[8 + t] = wmma16(a1, b[t & 3], acc[8 + t]);
    }
  }

  // ---- Epilogue: bias + sigmoid -> LDS (f16) ---------------------------
  // C/D layout (ISA 7.12.2): lane L, vgpr r -> (M = r + 8*(L>>4), N = L&15)
#pragma unroll
  for (int t = 0; t < 8; ++t) {
    int n = ncol[t];
    float bb = bias[n > (kInternal - 1) ? (kInternal - 1) : n];
#pragma unroll
    for (int h = 0; h < 2; ++h) {
#pragma unroll
      for (int r = 0; r < 8; ++r) {
        int m = r + 8 * kh + 16 * h;
        float v = acc[8 * h + t][r] + bb;
        float p = 1.0f / (1.0f + __expf(-v));
        s_probs[m * 1024 + n] = (_Float16)p;   // n==1023 is padding, never read
      }
    }
  }
  __syncthreads();

  // ---- Tree phase: 8 threads per row, 128 leaves each ------------------
  const int row = tid >> 3;              // 0..31
  const int sub = tid & 7;               // 0..7
  const _Float16* probs = s_probs + row * 1024;

  float c0 = 0.0f, c1 = 0.0f;
  for (int i = 0; i < 128; ++i) {
    int leaf = sub * 128 + i;
    int node = kInternal + leaf;
    float prod = 1.0f;
#pragma unroll
    for (int d = 0; d < 10; ++d) {
      int par = (node - 1) >> 1;
      float pr = (float)probs[par];
      prod *= (node & 1) ? (1.0f - pr) : pr;   // even child == right -> factor p
      node = par;
    }
    float l0 = leaf_logits[2 * leaf];
    float l1 = leaf_logits[2 * leaf + 1];
    float mx = fmaxf(l0, l1);
    float e0 = __expf(l0 - mx);
    float e1 = __expf(l1 - mx);
    float inv = 1.0f / (e0 + e1);
    c0 += prod * e0 * inv;
    c1 += prod * e1 * inv;
  }

  // ---- Reduce 8 partials per row ---------------------------------------
  s_red[tid * 2 + 0] = c0;
  s_red[tid * 2 + 1] = c1;
  __syncthreads();
  if (sub == 0) {
    float s0 = 0.0f, s1 = 0.0f;
#pragma unroll
    for (int i = 0; i < 8; ++i) {
      s0 += s_red[(row * 8 + i) * 2 + 0];
      s1 += s_red[(row * 8 + i) * 2 + 1];
    }
    out[(size_t)(m0 + row) * 2 + 0] = s0;
    out[(size_t)(m0 + row) * 2 + 1] = s1;
  }
}

extern "C" void kernel_launch(void* const* d_in, const int* in_sizes, int n_in,
                              void* d_out, int out_size, void* d_ws, size_t ws_size,
                              hipStream_t stream) {
  (void)in_sizes; (void)n_in; (void)out_size;
  const float* x    = (const float*)d_in[0];   // [16384, 4096]
  const float* w    = (const float*)d_in[1];   // [1023, 4096]
  const float* bias = (const float*)d_in[2];   // [1023]
  const float* leaf = (const float*)d_in[3];   // [1024, 2]
  float* out = (float*)d_out;                  // [16384, 2]

  dim3 block(256);
  dim3 grid(kBatch / kMTile);

  const size_t w_elems = (size_t)kInternal * kInDim;        // 4,190,208
  if (ws_size >= w_elems * sizeof(unsigned short)) {
    unsigned short* wh = (unsigned short*)d_ws;
    convert_w_f16<<<dim3((unsigned)(w_elems / 8 / 256)), block, 0, stream>>>(w, wh);
    sdt_fused_kernel<true><<<grid, block, 0, stream>>>(x, w, wh, bias, leaf, out);
  } else {
    sdt_fused_kernel<false><<<grid, block, 0, stream>>>(x, w, nullptr, bias, leaf, out);
  }
}
